// Linear3Bit_11450382811536
// MI455X (gfx1250) — matrix operand order repaired and checked by reading, not compiled, not run on hardware
//
#include <hip/hip_runtime.h>

typedef __attribute__((ext_vector_type(16))) _Float16 v16h;
typedef __attribute__((ext_vector_type(8)))  _Float16 v8h;
typedef __attribute__((ext_vector_type(4)))  _Float16 v4h;
typedef __attribute__((ext_vector_type(8)))  float    v8f;
typedef __attribute__((ext_vector_type(4)))  float    v4f;

typedef __attribute__((address_space(3))) const _Float16 lds_cf16;

#define OUT_F   4096
#define IN_F    4096
#define M_TOT   8192
#define BM      128
#define BN      128
#define BK      32
#define LDT     40          // BK + 8 halves of padding (80B row stride, 16B-aligned)
#define KT      (IN_F / BK) // 128 k-tiles

// wave-relative LDS byte offset (what VDST of async-to-LDS ops expects)
__device__ __forceinline__ unsigned lds_off(const _Float16* p) {
    return (unsigned)(size_t)(lds_cf16*)p;
}

__device__ __forceinline__ void async_ld_b128(unsigned lds_byte_off, const _Float16* g) {
    // GLOBAL_LOAD_ASYNC_TO_LDS_B128: per-lane 16B global->LDS, tracked by ASYNCcnt
    asm volatile("global_load_async_to_lds_b128 %0, %1, off"
                 :: "v"(lds_byte_off), "v"(g)
                 : "memory");
}

__device__ __forceinline__ void wait_asynccnt0() {
    asm volatile("s_wait_asynccnt 0x0" ::: "memory");
}

// ---------------------------------------------------------------------------
// Pass 1: 3-bit dequantization.  One thread per 3-byte triple -> 8 f16 weights.
// weight_q3 arrives as int32 holding one byte each (reference randint 0..255).
// ---------------------------------------------------------------------------
__global__ __launch_bounds__(256) void dequant3_kernel(
    const int* __restrict__ q3, const _Float16* __restrict__ params,
    _Float16* __restrict__ w, int ntriples)
{
    int t = blockIdx.x * 256 + threadIdx.x;
    if (t >= ntriples) return;
    int b0 = q3[3 * t + 0] & 255;
    int b1 = q3[3 * t + 1] & 255;
    int b2 = q3[3 * t + 2] & 255;
    int g  = t >> 3;                          // 8 triples per 64-weight group
    float lo = (float)params[2 * g + 0];
    float hi = (float)params[2 * g + 1];
    float sc = (hi - lo) * (1.0f / 7.0f);
    int q[8];
    q[0] =  b0       & 7;
    q[1] = (b0 >> 3) & 7;
    q[2] = ((b0 >> 6) & 3) | ((b1 & 1) << 2);
    q[3] = (b1 >> 1) & 7;
    q[4] = (b1 >> 4) & 7;
    q[5] = ((b1 >> 7) & 1) | ((b2 & 3) << 1);
    q[6] = (b2 >> 2) & 7;
    q[7] = (b2 >> 5) & 7;
    v8h out;
#pragma unroll
    for (int i = 0; i < 8; ++i)
        out[i] = (_Float16)fmaf((float)q[i], sc, lo);
    *(v8h*)(w + (size_t)t * 8) = out;
}

// ---------------------------------------------------------------------------
// Pass 2: WMMA GEMM  C[M,N] = X[M,K] * W[N,K]^T + bias.
// 256 threads = 8 wave32; block tile 128x128, K-step 32, double-buffered LDS.
// A (fp32 x) staged through VGPRs with cvt to f16; B (f16 w) staged with
// async global->LDS copies overlapped with the WMMA stream.
// ---------------------------------------------------------------------------
__global__ __launch_bounds__(256) void gemm_wmma_kernel(
    const float* __restrict__ X, const _Float16* __restrict__ W,
    const float* __restrict__ bias, float* __restrict__ Out)
{
    __shared__ __align__(16) _Float16 sA[2][BM * LDT];
    __shared__ __align__(16) _Float16 sB[2][BN * LDT];

    const int tid  = threadIdx.x;
    const int lane = tid & 31;
    const int wave = tid >> 5;
    const int wm   = wave >> 2;        // 0..1 : wave row (64 M-rows each)
    const int wn   = wave & 3;         // 0..3 : wave col (32 N-cols each)
    const int m0   = blockIdx.y * BM;
    const int n0   = blockIdx.x * BN;

    // fragment lane decomposition (CDNA5 16-bit WMMA layouts)
    const int fr   = lane & 15;        // row (A) / col (B) within 16
    const int hgrp = lane >> 4;        // lane half-group
    const int akh  = hgrp * 8;         // A: K chunk base {0,8}
    const int bkh  = hgrp * 8;         // B: K chunk base {0,8}

    // global staging decomposition
    const int arow = tid >> 3;         // 0..31  (x tile: 128x32 fp32, float4/thread)
    const int acol = (tid & 7) * 4;
    const int brow = tid >> 2;         // 0..63  (w tile: 128x32 f16, 16B/thread)
    const int bcol = (tid & 3) * 8;

    v8f zero = {};
    v8f acc[4][2];
#pragma unroll
    for (int i = 0; i < 4; ++i)
#pragma unroll
        for (int j = 0; j < 2; ++j) acc[i][j] = zero;

    auto loadA = [&](int kt, int st) {
        const float* src = X + (size_t)m0 * IN_F + kt * BK;
#pragma unroll
        for (int p = 0; p < 4; ++p) {
            int r = arow + p * 32;
            v4f d = *(const v4f*)(src + (size_t)r * IN_F + acol);
            v4h h;
            h[0] = (_Float16)d[0]; h[1] = (_Float16)d[1];
            h[2] = (_Float16)d[2]; h[3] = (_Float16)d[3];
            *(v4h*)(&sA[st][r * LDT + acol]) = h;
        }
    };
    auto loadB = [&](int kt, int st) {
        const _Float16* src = W + (size_t)n0 * IN_F + kt * BK + bcol;
#pragma unroll
        for (int p = 0; p < 2; ++p) {
            int r = brow + p * 64;
            async_ld_b128(lds_off(&sB[st][r * LDT + bcol]),
                          src + (size_t)r * IN_F);
        }
    };

    union Frag { v16h v; v8h h[2]; };
    // A 16x32 f16: lane holds row M=fr; halves 0..7 = K akh..akh+7, halves 8..15 = K akh+16..akh+23
    auto fragA = [&](int st, int i) -> v16h {
        const _Float16* p = &sA[st][(wm * 64 + i * 16 + fr) * LDT + akh];
        Frag f;
        f.h[0] = *(const v8h*)(p);
        f.h[1] = *(const v8h*)(p + 16);
        return f.v;
    };
    // B 32x16 f16: lane holds col N=fr; halves 0..7 = K bkh..bkh+7, halves 8..15 = K bkh+16..bkh+23
    auto fragB = [&](int st, int j) -> v16h {
        const _Float16* p = &sB[st][(wn * 32 + j * 16 + fr) * LDT + bkh];
        Frag f;
        f.h[0] = *(const v8h*)(p);
        f.h[1] = *(const v8h*)(p + 16);
        return f.v;
    };

    loadA(0, 0);
    loadB(0, 0);
    wait_asynccnt0();
    __syncthreads();

    int st = 0;
    for (int kt = 0; kt < KT; ++kt) {
        if (kt + 2 < KT) {  // gfx1250 global_prefetch_b8 two tiles ahead
            __builtin_prefetch(X + (size_t)(m0 + arow) * IN_F + (kt + 2) * BK + acol, 0, 0);
            __builtin_prefetch(W + (size_t)(n0 + brow) * IN_F + (kt + 2) * BK + bcol, 0, 0);
        }
        if (kt + 1 < KT) {
            loadA(kt + 1, st ^ 1);
            loadB(kt + 1, st ^ 1);
        }

        v16h a[4], b[2];
#pragma unroll
        for (int i = 0; i < 4; ++i) a[i] = fragA(st, i);
#pragma unroll
        for (int j = 0; j < 2; ++j) b[j] = fragB(st, j);

#pragma unroll
        for (int i = 0; i < 4; ++i)
#pragma unroll
            for (int j = 0; j < 2; ++j)
                acc[i][j] = __builtin_amdgcn_wmma_f32_16x16x32_f16(
                    /*neg_a=*/false, a[i], /*neg_b=*/false, b[j],
                    /*c_mod=*/(short)0, acc[i][j],
                    /*reuse_a=*/false, /*reuse_b=*/false);

        wait_asynccnt0();   // next-stage B tile landed (overlapped with WMMAs above)
        __syncthreads();
        st ^= 1;
    }

    // Epilogue: C layout -> VGPR v holds M = v + 8*hgrp, N = fr
#pragma unroll
    for (int i = 0; i < 4; ++i) {
        int rbase = m0 + wm * 64 + i * 16 + 8 * hgrp;
#pragma unroll
        for (int j = 0; j < 2; ++j) {
            int col  = n0 + wn * 32 + j * 16 + fr;
            float bv = bias[col];
            float* o = Out + (size_t)rbase * OUT_F + col;
#pragma unroll
            for (int v = 0; v < 8; ++v)
                o[(size_t)v * OUT_F] = acc[i][j][v] + bv;
        }
    }
}

// ---------------------------------------------------------------------------
extern "C" void kernel_launch(void* const* d_in, const int* in_sizes, int n_in,
                              void* d_out, int out_size, void* d_ws, size_t ws_size,
                              hipStream_t stream) {
    (void)in_sizes; (void)n_in; (void)out_size; (void)ws_size;
    const float*    x      = (const float*)d_in[0];
    const int*      q3     = (const int*)d_in[1];
    const _Float16* params = (const _Float16*)d_in[2];  // fp16 (lo,hi) pairs
    const float*    bias   = (const float*)d_in[3];
    float*          out    = (float*)d_out;
    _Float16*       wf16   = (_Float16*)d_ws;           // 4096*4096*2 = 32 MB scratch

    const int ngroups  = OUT_F * IN_F / 64;
    const int ntriples = ngroups * 8;   // 2,097,152 triples, 8 weights each
    dequant3_kernel<<<(ntriples + 255) / 256, 256, 0, stream>>>(q3, params, wf16, ntriples);

    dim3 grid(OUT_F / BN, M_TOT / BM);  // 32 x 64 blocks
    gemm_wmma_kernel<<<grid, 256, 0, stream>>>(x, wf16, bias, out);
}
